// TDPolicyNetwork_62242666053676
// MI455X (gfx1250) — matrix operand order repaired and checked by reading, not compiled, not run on hardware
//
#include <hip/hip_runtime.h>
#include <hip/hip_bf16.h>
#include <math.h>

typedef __attribute__((ext_vector_type(16))) _Float16 v16h;
typedef __attribute__((ext_vector_type(8)))  _Float16 v8h;
typedef __attribute__((ext_vector_type(8)))  float    v8f;

// problem dims
#define BB   8
#define SS   256
#define DD   16
#define HH   256
#define NHH  8
#define DHH  32
#define LLAY 3
#define FFF  512
#define NWG_LSTM 16

// ---------------------------------------------------------------------------
// WMMA fragment loaders (f16, 16x16x32)
// A-matrix 16x32: lane r(0-15),half: elems 0..7 -> K=half*8+j ; 8..15 -> K=16+half*8+j
// B-matrix 32x16: lane holds weight-row (out-col) n0+r, elems 0..7 -> K=half*8+j ; 8..15 -> K=16+half*8+j
// ---------------------------------------------------------------------------
__device__ __forceinline__ v16h frag_a(const _Float16* base, int ld, int r, int half, int k0) {
  const _Float16* p = base + (long)r * ld + k0 + half * 8;
  v8h lo = *(const v8h*)(p);
  v8h hi = *(const v8h*)(p + 16);
  v16h o;
#pragma unroll
  for (int i = 0; i < 8; ++i) { o[i] = lo[i]; o[i + 8] = hi[i]; }
  return o;
}

__device__ __forceinline__ v16h frag_b(const _Float16* rowbase, int k0, int half) {
  const _Float16* p = rowbase + k0 + half * 8;
  v8h lo = *(const v8h*)(p);
  v8h hi = *(const v8h*)(p + 16);
  v16h o;
#pragma unroll
  for (int i = 0; i < 8; ++i) { o[i] = lo[i]; o[i + 8] = hi[i]; }
  return o;
}

__device__ __forceinline__ float sigf(float x) { return 1.0f / (1.0f + __expf(-x)); }

// ---------------------------------------------------------------------------
// Generic batched WMMA GEMM, register-blocked MT x NT tiles per wave:
//   C[m,n] = alpha * sum_k A[m,k]*W[n,k] + bias[n]
// grid = (N/(16*NT), M/(16*MT), nz); z -> (z0=z/nz1, z1=z%nz1) with strides.
// ---------------------------------------------------------------------------
template <int MT, int NT>
__global__ __launch_bounds__(32) void gemm_wmma(
    const _Float16* __restrict__ A, const _Float16* __restrict__ Bw,
    const float* __restrict__ bias, float* __restrict__ C32, _Float16* __restrict__ C16,
    int K, int lda, int ldb, int ldc,
    long sA0, long sA1, long sB0, long sB1, long sC0, long sC1, int nz1,
    float alpha, int relu) {
  int n0 = blockIdx.x * (16 * NT);
  int m0 = blockIdx.y * (16 * MT);
  int z = blockIdx.z;
  int z0 = z / nz1, z1 = z - z0 * nz1;
  A  += z0 * sA0 + z1 * sA1;
  Bw += z0 * sB0 + z1 * sB1;
  long coff = z0 * sC0 + z1 * sC1;
  int lane = threadIdx.x;
  int r = lane & 15, half = lane >> 4;
  const _Float16* arow = A + (long)m0 * lda;
  const _Float16* brow = Bw + (long)(n0 + r) * ldb;
  v8f acc[MT][NT];
#pragma unroll
  for (int mi = 0; mi < MT; ++mi)
#pragma unroll
    for (int ni = 0; ni < NT; ++ni) acc[mi][ni] = (v8f){};
  for (int k0 = 0; k0 < K; k0 += 32) {
    v16h a[MT];
#pragma unroll
    for (int mi = 0; mi < MT; ++mi)
      a[mi] = frag_a(arow + (long)mi * 16 * lda, lda, r, half, k0);
#pragma unroll
    for (int ni = 0; ni < NT; ++ni) {
      v16h b = frag_b(brow + (long)ni * 16 * ldb, k0, half);
#pragma unroll
      for (int mi = 0; mi < MT; ++mi)
        acc[mi][ni] = __builtin_amdgcn_wmma_f32_16x16x32_f16(false, a[mi], false, b,
                                                             (short)0, acc[mi][ni], false, false);
    }
  }
#pragma unroll
  for (int ni = 0; ni < NT; ++ni) {
    int col = n0 + ni * 16 + r;
    float bv = bias ? bias[col] : 0.0f;
#pragma unroll
    for (int mi = 0; mi < MT; ++mi) {
#pragma unroll
      for (int i = 0; i < 8; ++i) {
        int row = m0 + mi * 16 + i + half * 8;
        float v = acc[mi][ni][i] * alpha + bv;
        if (relu) v = v > 0.0f ? v : 0.0f;
        long idx = coff + (long)row * ldc + col;
        if (C32) C32[idx] = v;
        if (C16) C16[idx] = (_Float16)v;
      }
    }
  }
}

// ---------------------------------------------------------------------------
// f32 -> f16 convert with K padding (zero-fill cols >= kin)
// ---------------------------------------------------------------------------
__global__ void cvt_pad(const float* __restrict__ src, _Float16* __restrict__ dst,
                        int rows, int kin, int kout) {
  long idx = (long)blockIdx.x * 256 + threadIdx.x;
  long total = (long)rows * kout;
  if (idx >= total) return;
  int r = (int)(idx / kout), c = (int)(idx % kout);
  dst[idx] = (c < kin) ? (_Float16)src[(long)r * kin + c] : (_Float16)0.0f;
}

// x[:,128:192] = pos_emb broadcast, x[:,192:256] = 0 (both f32 master + f16 copy)
__global__ void build_x_tail(const float* __restrict__ pos, float* __restrict__ x32,
                             _Float16* __restrict__ x16) {
  int idx = blockIdx.x * 256 + threadIdx.x;  // 2048*128 total
  int r = idx >> 7, j = idx & 127;
  int s = r & 255;
  float v = (j < 64) ? pos[s * 64 + j] : 0.0f;
  x32[(long)r * HH + 128 + j] = v;
  x16[(long)r * HH + 128 + j] = (_Float16)v;
}

// row softmax: scores f32 [rows][256] -> probs f16
__global__ void softmax_rows(const float* __restrict__ S, _Float16* __restrict__ P) {
  __shared__ float red[256];
  long row = blockIdx.x;
  int tid = threadIdx.x;
  float v = S[row * 256 + tid];
  red[tid] = v; __syncthreads();
  for (int s = 128; s > 0; s >>= 1) { if (tid < s) red[tid] = fmaxf(red[tid], red[tid + s]); __syncthreads(); }
  float m = red[0]; __syncthreads();
  float e = __expf(v - m);
  red[tid] = e; __syncthreads();
  for (int s = 128; s > 0; s >>= 1) { if (tid < s) red[tid] += red[tid + s]; __syncthreads(); }
  P[row * 256 + tid] = (_Float16)(e / red[0]);
}

// vT[b,h,d,s] = qkv16[(b*256+s)*768 + 512 + h*32 + d]
__global__ void transpose_v(const _Float16* __restrict__ qkv, _Float16* __restrict__ vt) {
  long idx = (long)blockIdx.x * 256 + threadIdx.x;  // 64*32*256 total
  int z = (int)(idx >> 13);          // b*8+h
  int rem = (int)(idx & 8191);
  int d = rem >> 8, s = rem & 255;
  int b = z >> 3, h = z & 7;
  vt[idx] = qkv[((long)(b * 256 + s)) * 768 + 512 + h * 32 + d];
}

// x = LN(x + res) * g + b ; writes f32 master and f16 copy
__global__ void ln_residual(float* __restrict__ x32, const float* __restrict__ res,
                            const float* __restrict__ g, const float* __restrict__ b,
                            _Float16* __restrict__ x16) {
  __shared__ float red[256];
  long row = blockIdx.x;
  int tid = threadIdx.x;
  float v = x32[row * HH + tid] + res[row * HH + tid];
  red[tid] = v; __syncthreads();
  for (int s = 128; s > 0; s >>= 1) { if (tid < s) red[tid] += red[tid + s]; __syncthreads(); }
  float mean = red[0] * (1.0f / HH); __syncthreads();
  float d = v - mean;
  red[tid] = d * d; __syncthreads();
  for (int s = 128; s > 0; s >>= 1) { if (tid < s) red[tid] += red[tid + s]; __syncthreads(); }
  float var = red[0] * (1.0f / HH);
  float o = d * rsqrtf(var + 1e-5f) * g[tid] + b[tid];
  x32[row * HH + tid] = o;
  x16[row * HH + tid] = (_Float16)o;
}

// dec_in16[t][row][h] : row<8 -> (t==0 ? 0 : enc16[row, ts[row][t-1], h]), rows 8..15 zero
__global__ void gather_decin(const _Float16* __restrict__ enc16, const int* __restrict__ ts,
                             _Float16* __restrict__ decin) {
  int blk = blockIdx.x;              // 256*16
  int t = blk >> 4, row = blk & 15;
  int h = threadIdx.x;
  _Float16 v = (_Float16)0.0f;
  if (t > 0 && row < 8) {
    int s = ts[row * SS + (t - 1)];
    v = enc16[((long)row * SS + s) * HH + h];
  }
  decin[((long)t * 16 + row) * HH + h] = v;
}

// zero the LSTM barrier counters and both h ping-pong buffers
__global__ void zero_lstm_state(unsigned* __restrict__ cnt, _Float16* __restrict__ hbuf) {
  int i = blockIdx.x * 256 + threadIdx.x;
  if (i < 256) cnt[i] = 0u;
  if (i < 2 * 16 * HH) hbuf[i] = (_Float16)0.0f;
}

// ---------------------------------------------------------------------------
// Multi-WGP LSTM: 16 single-wave workgroups. WG g owns h-columns 16g..16g+15,
// i.e. exactly 4 gate tiles (n, n+256, n+512, n+768): gate math + cell state
// stay in this wave's registers. Each WG preloads its 64 KB weight slice
// (Wih+Whh rows for its 64 gate columns) into LDS once, then all 256 steps
// read B-fragments from LDS (no L2 weight streaming, no register spills).
// Recurrence sync: double-buffered global h + per-step global counter barrier.
// ---------------------------------------------------------------------------
__global__ __launch_bounds__(32) void lstm_multi(
    const _Float16* __restrict__ decin,   // [256][16][256]
    const _Float16* __restrict__ Wih16,   // [1024][256]
    const _Float16* __restrict__ Whh16,   // [1024][256]
    const float* __restrict__ bih, const float* __restrict__ bhh,
    _Float16* __restrict__ hbuf,          // [2][16][256] ping-pong (pre-zeroed)
    unsigned* __restrict__ cnt,           // [256] (pre-zeroed)
    _Float16* __restrict__ decout) {      // [256][16][256]
  __shared__ _Float16 wsh[2 * 4 * 16 * HH];  // 64 KB: [mat][gate][row16][k256]
  int wg = blockIdx.x;                  // 0..15
  int lane = threadIdx.x;
  int r = lane & 15, half = lane >> 4;
  int colblk = wg * 16;
  // ---- preload weight slice into LDS ----
  for (int idx = lane; idx < 2 * 4 * 16 * HH; idx += 32) {
    int mat = idx >> 14;                // 0=Wih, 1=Whh
    int rem = idx & 16383;
    int q = rem >> 12;                  // gate 0..3
    int rr = (rem >> 8) & 15;           // row within tile
    int k = rem & 255;
    const _Float16* W = mat ? Whh16 : Wih16;
    wsh[idx] = W[((long)(q * 256 + colblk + rr)) * HH + k];
  }
  __syncthreads();
  int col = colblk + r;
  float bi_ = bih[col] + bhh[col];
  float bf_ = bih[256 + col] + bhh[256 + col];
  float bg_ = bih[512 + col] + bhh[512 + col];
  float bo_ = bih[768 + col] + bhh[768 + col];
  v8f c = {};                           // cell state: rows i+half*8, this col
  for (int t = 0; t < 256; ++t) {
    asm volatile("" ::: "memory");      // keep LDS B-frag loads inside the loop
    const _Float16* xt = decin + (long)t * 16 * HH;
    const _Float16* hprev = hbuf + (long)(t & 1) * 16 * HH;
    _Float16* hnext = hbuf + (long)((t + 1) & 1) * 16 * HH;
    v8f g4[4] = {{}, {}, {}, {}};
    for (int k0 = 0; k0 < HH; k0 += 32) {
      v16h a = frag_a(xt, HH, r, half, k0);
#pragma unroll
      for (int q = 0; q < 4; ++q) {
        v16h b = frag_b(&wsh[(q * 16 + r) * HH], k0, half);
        g4[q] = __builtin_amdgcn_wmma_f32_16x16x32_f16(false, a, false, b, (short)0, g4[q], false, false);
      }
    }
    for (int k0 = 0; k0 < HH; k0 += 32) {
      v16h a = frag_a(hprev, HH, r, half, k0);
#pragma unroll
      for (int q = 0; q < 4; ++q) {
        v16h b = frag_b(&wsh[(4 * 16 * HH) + (q * 16 + r) * HH], k0, half);
        g4[q] = __builtin_amdgcn_wmma_f32_16x16x32_f16(false, a, false, b, (short)0, g4[q], false, false);
      }
    }
#pragma unroll
    for (int i = 0; i < 8; ++i) {
      int row = i + half * 8;
      float iv = sigf(g4[0][i] + bi_);
      float fv = sigf(g4[1][i] + bf_);
      float gv = tanhf(g4[2][i] + bg_);
      float ov = sigf(g4[3][i] + bo_);
      float cn = fv * c[i] + iv * gv;
      c[i] = cn;
      float hv = ov * tanhf(cn);
      _Float16 hv16 = (_Float16)hv;
      hnext[row * HH + col] = hv16;
      decout[(long)t * (16 * HH) + row * HH + col] = hv16;
    }
    // global step barrier (release h stores, then spin until all 16 WGs arrive)
    __threadfence();
    if (lane == 0) {
      __hip_atomic_fetch_add(&cnt[t], 1u, __ATOMIC_RELEASE, __HIP_MEMORY_SCOPE_AGENT);
      while (__hip_atomic_load(&cnt[t], __ATOMIC_ACQUIRE, __HIP_MEMORY_SCOPE_AGENT) < (unsigned)NWG_LSTM) {}
    }
    __syncthreads();
  }
}

// logits[b,t,s] = sum_h v[h] * tanh(pd[b,t,h] + pen[b,s,h])
__global__ void pointer_kernel(const float* __restrict__ pd, const float* __restrict__ pen,
                               const float* __restrict__ vp, float* __restrict__ out) {
  __shared__ float pdr[256];
  __shared__ float vsh[256];
  int bt = blockIdx.x;                 // b*256 + t
  int b = bt >> 8, t = bt & 255;
  int s = threadIdx.x;
  pdr[s] = pd[((long)t * 16 + b) * HH + s];
  vsh[s] = vp[s];
  __syncthreads();
  const float* pr = pen + ((long)b * SS + s) * HH;
  float acc = 0.0f;
  for (int h = 0; h < HH; ++h) acc += vsh[h] * tanhf(pdr[h] + pr[h]);
  out[((long)b * SS + t) * SS + s] = acc;
}

// ---------------------------------------------------------------------------
// Host driver
// ---------------------------------------------------------------------------
static void launch_gemm(hipStream_t st, const _Float16* A, const _Float16* B, const float* bias,
                        float* C32, _Float16* C16, int M, int N, int K, int lda, int ldb, int ldc,
                        long sA0, long sA1, long sB0, long sB1, long sC0, long sC1,
                        int nz1, int nz, float alpha, int relu) {
  if ((M % 32 == 0) && (N % 64 == 0)) {
    dim3 g(N / 64, M / 32, nz);
    gemm_wmma<2, 4><<<g, dim3(32), 0, st>>>(A, B, bias, C32, C16, K, lda, ldb, ldc,
                                            sA0, sA1, sB0, sB1, sC0, sC1, nz1, alpha, relu);
  } else if ((M % 32 == 0) && (N % 32 == 0)) {
    dim3 g(N / 32, M / 32, nz);
    gemm_wmma<2, 2><<<g, dim3(32), 0, st>>>(A, B, bias, C32, C16, K, lda, ldb, ldc,
                                            sA0, sA1, sB0, sB1, sC0, sC1, nz1, alpha, relu);
  } else {
    dim3 g(N / 16, M / 16, nz);
    gemm_wmma<1, 1><<<g, dim3(32), 0, st>>>(A, B, bias, C32, C16, K, lda, ldb, ldc,
                                            sA0, sA1, sB0, sB1, sC0, sC1, nz1, alpha, relu);
  }
}

extern "C" void kernel_launch(void* const* d_in, const int* in_sizes, int n_in,
                              void* d_out, int out_size, void* d_ws, size_t ws_size,
                              hipStream_t stream) {
  (void)in_sizes; (void)n_in; (void)out_size; (void)ws_size;
  const float* parts   = (const float*)d_in[0];
  const int*   tseq    = (const int*)d_in[1];
  const float* W_pe1   = (const float*)d_in[2];
  const float* b_pe1   = (const float*)d_in[3];
  const float* W_pe2   = (const float*)d_in[4];
  const float* b_pe2   = (const float*)d_in[5];
  const float* pos_emb = (const float*)d_in[6];
  const float* Wqkv    = (const float*)d_in[7];
  const float* bqkv    = (const float*)d_in[8];
  const float* Wo      = (const float*)d_in[9];
  const float* bo      = (const float*)d_in[10];
  const float* ln1_g   = (const float*)d_in[11];
  const float* ln1_b   = (const float*)d_in[12];
  const float* W1f     = (const float*)d_in[13];
  const float* b1f     = (const float*)d_in[14];
  const float* W2f     = (const float*)d_in[15];
  const float* b2f     = (const float*)d_in[16];
  const float* ln2_g   = (const float*)d_in[17];
  const float* ln2_b   = (const float*)d_in[18];
  const float* Wih     = (const float*)d_in[19];
  const float* Whh     = (const float*)d_in[20];
  const float* bih     = (const float*)d_in[21];
  const float* bhh     = (const float*)d_in[22];
  const float* Wp      = (const float*)d_in[23];
  const float* bp      = (const float*)d_in[24];
  const float* v_ptr   = (const float*)d_in[25];
  float* out = (float*)d_out;

  // ---- workspace carve (~51 MB) ----
  char* base = (char*)d_ws;
  size_t off = 0;
  auto carve = [&](size_t bytes) { char* p = base + off; off = (off + bytes + 255) & ~(size_t)255; return p; };
  _Float16* Wpe1h = (_Float16*)carve((size_t)128 * 32 * 2);
  _Float16* Wpe2h = (_Float16*)carve((size_t)128 * 128 * 2);
  _Float16* Wqkvh = (_Float16*)carve((size_t)3 * 768 * 256 * 2);
  _Float16* Woh   = (_Float16*)carve((size_t)3 * 256 * 256 * 2);
  _Float16* W1fh  = (_Float16*)carve((size_t)3 * 512 * 256 * 2);
  _Float16* W2fh  = (_Float16*)carve((size_t)3 * 256 * 512 * 2);
  _Float16* Wihh  = (_Float16*)carve((size_t)1024 * 256 * 2);
  _Float16* Whhh  = (_Float16*)carve((size_t)1024 * 256 * 2);
  _Float16* Wph   = (_Float16*)carve((size_t)256 * 256 * 2);
  _Float16* parts16 = (_Float16*)carve((size_t)2048 * 32 * 2);
  _Float16* t1h   = (_Float16*)carve((size_t)2048 * 128 * 2);
  float*    x32   = (float*)carve((size_t)2048 * 256 * 4);
  _Float16* x16   = (_Float16*)carve((size_t)2048 * 256 * 2);
  _Float16* qkv16 = (_Float16*)carve((size_t)2048 * 768 * 2);
  float*    scores = (float*)carve((size_t)64 * 256 * 256 * 4);
  _Float16* probs16 = (_Float16*)carve((size_t)64 * 256 * 256 * 2);
  _Float16* vt16  = (_Float16*)carve((size_t)64 * 32 * 256 * 2);
  _Float16* o16   = (_Float16*)carve((size_t)2048 * 256 * 2);
  float*    resid = (float*)carve((size_t)2048 * 256 * 4);
  _Float16* f1h   = (_Float16*)carve((size_t)2048 * 512 * 2);
  _Float16* decin = (_Float16*)carve((size_t)256 * 16 * 256 * 2);
  _Float16* decout = (_Float16*)carve((size_t)256 * 16 * 256 * 2);
  float*    pd    = (float*)carve((size_t)4096 * 256 * 4);
  float*    pen   = (float*)carve((size_t)2048 * 256 * 4);
  _Float16* hbuf  = (_Float16*)carve((size_t)2 * 16 * 256 * 2);
  unsigned* cnt   = (unsigned*)carve((size_t)256 * 4);

  auto cvt = [&](const float* s, _Float16* d, int rows, int kin, int kout) {
    long total = (long)rows * kout;
    cvt_pad<<<dim3((total + 255) / 256), dim3(256), 0, stream>>>(s, d, rows, kin, kout);
  };
  // ---- weight / input conversion to f16 (K-padded where needed) ----
  cvt(W_pe1, Wpe1h, 128, 16, 32);
  cvt(W_pe2, Wpe2h, 128, 128, 128);
  cvt(Wqkv,  Wqkvh, 3 * 768, 256, 256);
  cvt(Wo,    Woh,   3 * 256, 256, 256);
  cvt(W1f,   W1fh,  3 * 512, 256, 256);
  cvt(W2f,   W2fh,  3 * 256, 512, 512);
  cvt(Wih,   Wihh,  1024, 256, 256);
  cvt(Whh,   Whhh,  1024, 256, 256);
  cvt(Wp,    Wph,   256, 256, 256);
  cvt(parts, parts16, 2048, 16, 32);

  // ---- part encoder: relu(parts@Wpe1^T+b)@Wpe2^T+b -> x[:, :128]; tail = pos|0 ----
  launch_gemm(stream, parts16, Wpe1h, b_pe1, nullptr, t1h, 2048, 128, 32, 32, 32, 128,
              0, 0, 0, 0, 0, 0, 1, 1, 1.0f, 1);
  launch_gemm(stream, t1h, Wpe2h, b_pe2, x32, x16, 2048, 128, 128, 128, 128, 256,
              0, 0, 0, 0, 0, 0, 1, 1, 1.0f, 0);
  build_x_tail<<<dim3(1024), dim3(256), 0, stream>>>(pos_emb, x32, x16);

  // ---- transformer encoder, 3 layers ----
  const float scale = 0.17677669529663687f;  // 1/sqrt(32)
  for (int l = 0; l < LLAY; ++l) {
    // QKV projection -> qkv16 [2048,768]
    launch_gemm(stream, x16, Wqkvh + (size_t)l * 768 * 256, bqkv + l * 768, nullptr, qkv16,
                2048, 768, 256, 256, 256, 768, 0, 0, 0, 0, 0, 0, 1, 1, 1.0f, 0);
    // scores[b,h] = q @ k^T * scale   (z = b*8+h)
    launch_gemm(stream, qkv16, qkv16 + 256, nullptr, scores, nullptr,
                256, 256, 32, 768, 768, 256,
                (long)256 * 768, 32, (long)256 * 768, 32,
                (long)8 * 65536, 65536, 8, 64, scale, 0);
    softmax_rows<<<dim3(64 * 256), dim3(256), 0, stream>>>(scores, probs16);
    transpose_v<<<dim3(64 * 32 * 256 / 256), dim3(256), 0, stream>>>(qkv16, vt16);
    // o[b,h] = probs @ v  -> o16 columns h*32..h*32+31
    launch_gemm(stream, probs16, vt16, nullptr, nullptr, o16,
                256, 32, 256, 256, 256, 256,
                (long)8 * 65536, 65536, (long)8 * 8192, 8192,
                (long)256 * 256, 32, 8, 64, 1.0f, 0);
    // output projection -> resid
    launch_gemm(stream, o16, Woh + (size_t)l * 65536, bo + l * 256, resid, nullptr,
                2048, 256, 256, 256, 256, 256, 0, 0, 0, 0, 0, 0, 1, 1, 1.0f, 0);
    ln_residual<<<dim3(2048), dim3(256), 0, stream>>>(x32, resid, ln1_g + l * 256, ln1_b + l * 256, x16);
    // FFN
    launch_gemm(stream, x16, W1fh + (size_t)l * 512 * 256, b1f + l * 512, nullptr, f1h,
                2048, 512, 256, 256, 256, 512, 0, 0, 0, 0, 0, 0, 1, 1, 1.0f, 1);
    launch_gemm(stream, f1h, W2fh + (size_t)l * 256 * 512, b2f + l * 256, resid, nullptr,
                2048, 256, 512, 512, 512, 256, 0, 0, 0, 0, 0, 0, 1, 1, 1.0f, 0);
    ln_residual<<<dim3(2048), dim3(256), 0, stream>>>(x32, resid, ln2_g + l * 256, ln2_b + l * 256, x16);
  }
  // x16 is now enc (f16), x32 enc (f32)

  // ---- teacher-forced decoder input gather, then multi-WGP LSTM ----
  gather_decin<<<dim3(256 * 16), dim3(256), 0, stream>>>(x16, tseq, decin);
  zero_lstm_state<<<dim3(32), dim3(256), 0, stream>>>(cnt, hbuf);
  lstm_multi<<<dim3(NWG_LSTM), dim3(32), 0, stream>>>(decin, Wihh, Whhh, bih, bhh, hbuf, cnt, decout);

  // ---- pointer head ----
  launch_gemm(stream, decout, Wph, bp, pd, nullptr, 4096, 256, 256, 256, 256, 256,
              0, 0, 0, 0, 0, 0, 1, 1, 1.0f, 0);
  launch_gemm(stream, x16, Wph, bp, pen, nullptr, 2048, 256, 256, 256, 256, 256,
              0, 0, 0, 0, 0, 0, 1, 1, 1.0f, 0);
  pointer_kernel<<<dim3(BB * SS), dim3(256), 0, stream>>>(pd, pen, v_ptr, out);
}
